// BahdanauDecoder_76338748719825
// MI455X (gfx1250) — matrix operand order repaired and checked by reading, not compiled, not run on hardware
//
#include <hip/hip_runtime.h>
#include <hip/hip_bf16.h>
#include <stdint.h>

// Problem constants (from reference)
#define B_  64
#define S_  128
#define T_  128
#define E_  512
#define H_  1024
#define NEG_ (-1e9f)

typedef __bf16 bf16_t;
typedef __attribute__((ext_vector_type(16))) __bf16 bfx16;
typedef __attribute__((ext_vector_type(8)))  float  fx8;

// ---------------------------------------------------------------------------
// fp32 -> bf16 (round to nearest even)
__device__ __forceinline__ unsigned short f32_to_bf16_bits(float f) {
    unsigned int u = __float_as_uint(f);
    unsigned int r = u + 0x7FFFu + ((u >> 16) & 1u);
    return (unsigned short)(r >> 16);
}

__global__ void convert_f32_bf16(const float* __restrict__ src,
                                 unsigned short* __restrict__ dst, int n) {
    int i = blockIdx.x * blockDim.x + threadIdx.x;
    if (i < n) dst[i] = f32_to_bf16_bits(src[i]);
}

__global__ void init_hidden(const float* __restrict__ encf,
                            float* __restrict__ h,
                            unsigned short* __restrict__ hb, int n) {
    int i = blockIdx.x * blockDim.x + threadIdx.x;
    if (i < n) { float v = encf[i]; h[i] = v; hb[i] = f32_to_bf16_bits(v); }
}

// ---------------------------------------------------------------------------
// Multi-tile WMMA bf16 GEMM device body:
//   C(M,N) = A(M,K) @ W(N,K)^T [+ bias(N)],  M a multiple of 64.
// Each wave owns a 64-row x 16-col strip: per k-step it loads ONE B (weight)
// fragment and FOUR A fragments, then issues 4 independent WMMAs (shared B,
// 4 accumulators). W row-major (N,K) == column-major B(K,N): per-lane 16
// contiguous K values -> one 32-byte run per B fragment.
__device__ __forceinline__ void gemm_strip_bf16(const bf16_t* __restrict__ A,
                                                const bf16_t* __restrict__ W,
                                                const float* __restrict__ bias,
                                                float* __restrict__ C,
                                                int N, int K,
                                                int mgroups, int wid, int lane) {
    int mg = wid % mgroups;            // 64-row group
    int nt = wid / mgroups;            // 16-col tile
    int m0 = mg << 6, n0 = nt << 4;

    // A fragment addressing (contiguous k = 16h + s, matching B)
    int abase = (lane < 16) ? 0 : 16;
    const bf16_t* ap0 = A + (size_t)(m0 +  0 + (lane & 15)) * K;
    const bf16_t* ap1 = A + (size_t)(m0 + 16 + (lane & 15)) * K;
    const bf16_t* ap2 = A + (size_t)(m0 + 32 + (lane & 15)) * K;
    const bf16_t* ap3 = A + (size_t)(m0 + 48 + (lane & 15)) * K;

    // B fragment addressing (contiguous k = 16h + s, matching A)
    int bbase = (lane < 16) ? 0 : 16;
    const bf16_t* bp = W + (size_t)(n0 + (lane & 15)) * K + bbase;

    fx8 c0 = {0.f,0.f,0.f,0.f,0.f,0.f,0.f,0.f};
    fx8 c1 = c0, c2 = c0, c3 = c0;

    for (int k0 = 0; k0 < K; k0 += 32) {
        if (k0 + 32 < K) {
            __builtin_prefetch(bp  + k0 + 32, 0, 1);   // global_prefetch_b8
            __builtin_prefetch(ap0 + k0 + 32, 0, 1);
        }
        bfx16 b, a0, a1, a2, a3;
#pragma unroll
        for (int j = 0; j < 16; ++j) b[j] = bp[k0 + j];
#pragma unroll
        for (int j = 0; j < 8; ++j) {
            a0[j] = ap0[k0 + abase + j];  a0[8 + j] = ap0[k0 + 8 + abase + j];
            a1[j] = ap1[k0 + abase + j];  a1[8 + j] = ap1[k0 + 8 + abase + j];
            a2[j] = ap2[k0 + abase + j];  a2[8 + j] = ap2[k0 + 8 + abase + j];
            a3[j] = ap3[k0 + abase + j];  a3[8 + j] = ap3[k0 + 8 + abase + j];
        }
        c0 = __builtin_amdgcn_wmma_f32_16x16x32_bf16(false, a0, false, b, (short)0, c0, false, false);
        c1 = __builtin_amdgcn_wmma_f32_16x16x32_bf16(false, a1, false, b, (short)0, c1, false, false);
        c2 = __builtin_amdgcn_wmma_f32_16x16x32_bf16(false, a2, false, b, (short)0, c2, false, false);
        c3 = __builtin_amdgcn_wmma_f32_16x16x32_bf16(false, a3, false, b, (short)0, c3, false, false);
    }

    int col   = n0 + (lane & 15);
    int rbase = (lane < 16) ? 0 : 8;
    float bv  = bias ? bias[col] : 0.f;
#pragma unroll
    for (int r = 0; r < 8; ++r) {
        C[(size_t)(m0 +  0 + rbase + r) * N + col] = c0[r] + bv;
        C[(size_t)(m0 + 16 + rbase + r) * N + col] = c1[r] + bv;
        C[(size_t)(m0 + 32 + rbase + r) * N + col] = c2[r] + bv;
        C[(size_t)(m0 + 48 + rbase + r) * N + col] = c3[r] + bv;
    }
}

// Single GEMM kernel (used for proj_key and gx)
__global__ void wmma_gemm_bf16(const bf16_t* __restrict__ A,
                               const bf16_t* __restrict__ W,
                               const float* __restrict__ bias,
                               float* __restrict__ C,
                               int N, int K, int mgroups) {
    int wid  = (blockIdx.x * blockDim.x + threadIdx.x) >> 5;
    int lane = threadIdx.x & 31;
    if (wid >= mgroups * (N >> 4)) return;     // wave-uniform guard
    gemm_strip_bf16(A, W, bias, C, N, K, mgroups, wid, lane);
}

// Dual GEMM kernel: waves [0, waves0) run GEMM-0 (q = h@Wq^T, no bias),
// waves [waves0, waves0+waves1) run GEMM-1 (gh = h@Whh^T + bhh).
// Both depend only on h, so they share one launch on the serial chain.
__global__ void wmma_dual_gemm_bf16(const bf16_t* __restrict__ A0,
                                    const bf16_t* __restrict__ W0,
                                    float* __restrict__ C0, int N0, int K0, int mg0,
                                    const bf16_t* __restrict__ A1,
                                    const bf16_t* __restrict__ W1,
                                    const float* __restrict__ bias1,
                                    float* __restrict__ C1, int N1, int K1, int mg1) {
    int wid  = (blockIdx.x * blockDim.x + threadIdx.x) >> 5;
    int lane = threadIdx.x & 31;
    int waves0 = mg0 * (N0 >> 4);
    int waves1 = mg1 * (N1 >> 4);
    if (wid < waves0) {
        gemm_strip_bf16(A0, W0, nullptr, C0, N0, K0, mg0, wid, lane);
    } else if (wid < waves0 + waves1) {
        gemm_strip_bf16(A1, W1, bias1, C1, N1, K1, mg1, wid - waves0, lane);
    }
}

// ---------------------------------------------------------------------------
// Fused attention step: e = tanh(q + proj_key) . v_att ; mask ; softmax ;
// ctx = alphas @ enc ; xcat = bf16([x_t | ctx]).  One block per batch row.
__global__ void attn_step(const float* __restrict__ q,          // (B,H)
                          const float* __restrict__ proj_key,   // (B,S,H)
                          const float* __restrict__ enc,        // (B,S,H)
                          const float* __restrict__ vatt,       // (H)
                          const unsigned char* __restrict__ mask,// (B,S)
                          const float* __restrict__ inputs,     // (B,T,E)
                          unsigned short* __restrict__ xcat,    // (B,E+H) bf16
                          int t) {
    __shared__ float q_s[H_];
    __shared__ float v_s[H_];
    __shared__ float e_s[S_];

    int b   = blockIdx.x;
    int tid = threadIdx.x;                 // 256 threads = 8 waves
    int wave = tid >> 5, lane = tid & 31;

    for (int i = tid; i < H_; i += 256) { q_s[i] = q[b * H_ + i]; v_s[i] = vatt[i]; }
    __syncthreads();

    const float* pk = proj_key + (size_t)b * S_ * H_;
    for (int s = wave; s < S_; s += 8) {
        const float* row = pk + (size_t)s * H_;
        float acc = 0.f;
        for (int h = lane; h < H_; h += 32)
            acc += tanhf(q_s[h] + row[h]) * v_s[h];
#pragma unroll
        for (int off = 16; off > 0; off >>= 1) acc += __shfl_xor(acc, off, 32);
        if (lane == 0) e_s[s] = mask[b * S_ + s] ? acc : NEG_;
    }
    __syncthreads();

    if (wave == 0) {   // softmax over S=128 by one wave
        float e0 = e_s[lane], e1 = e_s[lane + 32], e2 = e_s[lane + 64], e3 = e_s[lane + 96];
        float m = fmaxf(fmaxf(e0, e1), fmaxf(e2, e3));
#pragma unroll
        for (int off = 16; off > 0; off >>= 1) m = fmaxf(m, __shfl_xor(m, off, 32));
        float p0 = __expf(e0 - m), p1 = __expf(e1 - m), p2 = __expf(e2 - m), p3 = __expf(e3 - m);
        float sum = p0 + p1 + p2 + p3;
#pragma unroll
        for (int off = 16; off > 0; off >>= 1) sum += __shfl_xor(sum, off, 32);
        float inv = 1.f / sum;
        e_s[lane] = p0 * inv; e_s[lane + 32] = p1 * inv;
        e_s[lane + 64] = p2 * inv; e_s[lane + 96] = p3 * inv;
    }
    __syncthreads();

    // ctx: 4 consecutive h per thread (vectorized loads along H)
    const float* eb = enc + (size_t)b * S_ * H_;
    int h0 = tid * 4;
    float4 acc = make_float4(0.f, 0.f, 0.f, 0.f);
    for (int s = 0; s < S_; ++s) {
        float al = e_s[s];
        const float4 ev = *(const float4*)(eb + (size_t)s * H_ + h0);
        acc.x += al * ev.x; acc.y += al * ev.y;
        acc.z += al * ev.z; acc.w += al * ev.w;
    }
    unsigned short* xo = xcat + (size_t)b * (E_ + H_);
    xo[E_ + h0 + 0] = f32_to_bf16_bits(acc.x);
    xo[E_ + h0 + 1] = f32_to_bf16_bits(acc.y);
    xo[E_ + h0 + 2] = f32_to_bf16_bits(acc.z);
    xo[E_ + h0 + 3] = f32_to_bf16_bits(acc.w);

    const float* xin = inputs + (size_t)b * T_ * E_ + (size_t)t * E_;
    for (int i = tid; i < E_; i += 256) xo[i] = f32_to_bf16_bits(xin[i]);
}

// ---------------------------------------------------------------------------
// GRU gate fusion: r/z/n + hidden update; writes fp32 h, bf16 h, outputs[:,t,:]
__global__ void gru_gates(const float* __restrict__ gx,   // (B,3H)
                          const float* __restrict__ gh,   // (B,3H)
                          float* __restrict__ h,          // (B,H)
                          unsigned short* __restrict__ hb,// (B,H) bf16
                          float* __restrict__ outs,       // (B,T,H)
                          int t) {
    int j = blockIdx.x * blockDim.x + threadIdx.x;  // 0 .. B*H-1
    int b = j >> 10, hc = j & (H_ - 1);
    const float* gxr = gx + (size_t)b * 3 * H_;
    const float* ghr = gh + (size_t)b * 3 * H_;
    float r  = 1.f / (1.f + __expf(-(gxr[hc] + ghr[hc])));
    float z  = 1.f / (1.f + __expf(-(gxr[H_ + hc] + ghr[H_ + hc])));
    float n  = tanhf(gxr[2 * H_ + hc] + r * ghr[2 * H_ + hc]);
    float hp = h[j];
    float hn = (1.f - z) * n + z * hp;
    h[j]  = hn;
    hb[j] = f32_to_bf16_bits(hn);
    outs[((size_t)b * T_ + t) * H_ + hc] = hn;
}

// ---------------------------------------------------------------------------
extern "C" void kernel_launch(void* const* d_in, const int* in_sizes, int n_in,
                              void* d_out, int out_size, void* d_ws, size_t ws_size,
                              hipStream_t stream) {
    const float*         inputs = (const float*)d_in[0];          // (B,T,E)
    const float*         enc    = (const float*)d_in[1];          // (B,S,H)
    const float*         encf   = (const float*)d_in[2];          // (1,B,H)
    const unsigned char* mask   = (const unsigned char*)d_in[3];  // (B,S) bool
    const float*         Wkey   = (const float*)d_in[4];          // (H,H)
    const float*         Wq     = (const float*)d_in[5];          // (H,H)
    const float*         vatt   = (const float*)d_in[6];          // (H)
    const float*         Wih    = (const float*)d_in[7];          // (3H,E+H)
    const float*         Whh    = (const float*)d_in[8];          // (3H,H)
    const float*         bih    = (const float*)d_in[9];          // (3H)
    const float*         bhh    = (const float*)d_in[10];         // (3H)

    // ---- workspace carve-up (256B aligned) ----
    char* ws = (char*)d_ws;
    size_t off = 0;
    auto carve = [&](size_t bytes) -> char* {
        char* p = ws + off;
        off = (off + bytes + 255) & ~(size_t)255;
        return p;
    };
    unsigned short* enc_b  = (unsigned short*)carve((size_t)B_ * S_ * H_ * 2);
    unsigned short* Wkey_b = (unsigned short*)carve((size_t)H_ * H_ * 2);
    unsigned short* Wq_b   = (unsigned short*)carve((size_t)H_ * H_ * 2);
    unsigned short* Wih_b  = (unsigned short*)carve((size_t)3 * H_ * (E_ + H_) * 2);
    unsigned short* Whh_b  = (unsigned short*)carve((size_t)3 * H_ * H_ * 2);
    float*          pkey   = (float*)carve((size_t)B_ * S_ * H_ * 4);
    float*          h_f    = (float*)carve((size_t)B_ * H_ * 4);
    unsigned short* h_b    = (unsigned short*)carve((size_t)B_ * H_ * 2);
    float*          q_f    = (float*)carve((size_t)B_ * H_ * 4);
    unsigned short* xcat   = (unsigned short*)carve((size_t)B_ * (E_ + H_) * 2);
    float*          gx     = (float*)carve((size_t)B_ * 3 * H_ * 4);
    float*          gh     = (float*)carve((size_t)B_ * 3 * H_ * 4);
    (void)ws_size; (void)in_sizes; (void)n_in; (void)out_size;

    float* out_hidden = (float*)d_out;              // (1,B,H)
    float* out_seq    = (float*)d_out + B_ * H_;    // (B,T,H)

    auto cvt = [&](const float* s, unsigned short* d, int n) {
        convert_f32_bf16<<<(n + 255) / 256, 256, 0, stream>>>(s, d, n);
    };

    // ---- one-time setup ----
    cvt(enc,  enc_b,  B_ * S_ * H_);
    cvt(Wkey, Wkey_b, H_ * H_);
    cvt(Wq,   Wq_b,   H_ * H_);
    cvt(Wih,  Wih_b,  3 * H_ * (E_ + H_));
    cvt(Whh,  Whh_b,  3 * H_ * H_);
    init_hidden<<<(B_ * H_ + 255) / 256, 256, 0, stream>>>(encf, h_f, h_b, B_ * H_);

    auto gemm = [&](const unsigned short* A, const unsigned short* W,
                    const float* bias, float* C, int M, int N, int K) {
        int mgroups = M >> 6;                   // 64-row strips
        int waves   = mgroups * (N >> 4);
        int threads = waves * 32;
        wmma_gemm_bf16<<<(threads + 255) / 256, 256, 0, stream>>>(
            (const bf16_t*)A, (const bf16_t*)W, bias, C, N, K, mgroups);
    };

    // proj_key = enc @ W_key^T : (B*S, H) x (H, H)
    gemm(enc_b, Wkey_b, nullptr, pkey, B_ * S_, H_, H_);

    // dual-GEMM launch geometry: q (64 waves) + gh (192 waves)
    const int waves_q  = (B_ >> 6) * (H_ >> 4);
    const int waves_gh = (B_ >> 6) * ((3 * H_) >> 4);
    const int dual_threads = (waves_q + waves_gh) * 32;

    // ---- sequential scan over T ----
    for (int t = 0; t < T_; ++t) {
        // q = h @ W_query^T  AND  gh = h @ W_hh^T + b_hh  (one launch)
        wmma_dual_gemm_bf16<<<(dual_threads + 255) / 256, 256, 0, stream>>>(
            (const bf16_t*)h_b, (const bf16_t*)Wq_b, q_f, H_, H_, B_ >> 6,
            (const bf16_t*)h_b, (const bf16_t*)Whh_b, bhh, gh, 3 * H_, H_, B_ >> 6);
        // attention + build bf16 [x_t | ctx]
        attn_step<<<B_, 256, 0, stream>>>(q_f, pkey, enc, vatt, mask,
                                          inputs, xcat, t);
        // gx = [x_t|ctx] @ W_ih^T + b_ih
        gemm(xcat, Wih_b, bih, gx, B_, 3 * H_, E_ + H_);
        // gate fusion + hidden update + write outputs[:, t, :]
        gru_gates<<<(B_ * H_) / 256, 256, 0, stream>>>(gx, gh, h_f, h_b,
                                                       out_seq, t);
    }

    // final hidden (1,B,H)
    hipMemcpyAsync(out_hidden, h_f, (size_t)B_ * H_ * sizeof(float),
                   hipMemcpyDeviceToDevice, stream);
}
